// Pair_update_42528766165482
// MI455X (gfx1250) — matrix operand order repaired and checked by reading, not compiled, not run on hardware
//
#include <hip/hip_runtime.h>

// ---------------------------------------------------------------------------
// RoseTTAFold-style pair update block for MI455X (gfx1250, wave32, WMMA).
// Heavy contractions: v_wmma_f32_16x16x32_bf16 (bf16 in, fp32 accum).
// LDS staging: global_load_async_to_lds_b128 (ASYNCcnt), double-buffered in
// the tied-logits K-loop so copies overlap the matrix pipe.
// ---------------------------------------------------------------------------

#define LSEQ 384
#define LL   (LSEQ*LSEQ)          // 147456 rows
#define CDIM 128
#define NHEAD 8
#define DHID 16
#define SCALE_Q 0.25f             // 1/sqrt(16)
#define INV_L (1.0f/384.0f)

typedef __attribute__((ext_vector_type(16))) __bf16 v16bf;
typedef __attribute__((ext_vector_type(8)))  float  v8f;

static __device__ __forceinline__ float bf2f(unsigned short u) {
  union { float f; unsigned int i; } x; x.i = ((unsigned int)u) << 16; return x.f;
}
static __device__ __forceinline__ unsigned short f2bf(float f) {
  union { float f; unsigned int i; } x; x.f = f;
  unsigned int r = x.i + 0x7fffu + ((x.i >> 16) & 1u);   // round-to-nearest-even
  return (unsigned short)(r >> 16);
}

union FragBF { uint4 q[2]; v16bf v; };
union AccU   { v8f v; float f[8]; };

// A-operand fragment (16x32 bf16, M = lane%16, K = (lane/16)*16 + e sequential):
// lane<16 holds K {0..15}, lane>=16 holds K {16..31}; each half
// range is 16B contiguous -> two ds_load_b128 per lane.
static __device__ __forceinline__ v16bf load_fragA(const unsigned short* row, int lane) {
  const int g = (lane & 16) ? 16 : 0;
  FragBF f;
  f.q[0] = *(const uint4*)(row + g);
  f.q[1] = *(const uint4*)(row + g + 8);
  return f.v;
}
// B-operand fragment (32x16 bf16, N = lane%16, K = (lane/16)*16 + e sequential)
static __device__ __forceinline__ v16bf load_fragB(const unsigned short* row, int lane) {
  const int kb = (lane & 16) ? 16 : 0;
  FragBF f;
  f.q[0] = *(const uint4*)(row + kb);
  f.q[1] = *(const uint4*)(row + kb + 8);
  return f.v;
}
static __device__ __forceinline__ v8f wmma_bf16(v16bf a, v16bf b, v8f c) {
  return __builtin_amdgcn_wmma_f32_16x16x32_bf16(false, a, false, b, (short)0, c, false, false);
}

// async copy of one 16B chunk per lane into LDS (ASYNCcnt-tracked)
static __device__ __forceinline__ void async_b128(unsigned lds_off, const void* gaddr) {
  asm volatile("global_load_async_to_lds_b128 %0, %1, off"
               :: "v"(lds_off), "v"(gaddr) : "memory");
}
static __device__ __forceinline__ unsigned lds_off32(const void* p) {
  return (unsigned)(size_t)p;    // LDS aperture: low 32 bits = LDS byte offset
}

// ---------------------------------------------------------------------------
// 1. node projection: left = node @ W[:256], right = node @ W[256:]
// ---------------------------------------------------------------------------
__global__ void node_proj_kernel(const float* __restrict__ node,
                                 const float* __restrict__ proj_w,
                                 float* __restrict__ left, float* __restrict__ right) {
  int t = blockIdx.x * 256 + threadIdx.x;       // 384*256 threads
  if (t >= LSEQ * 256) return;
  int i = t >> 8, c2 = t & 255;
  int half = c2 >> 7, c = c2 & 127;
  const float* nrow = node + (size_t)i * 256;
  const float* w = proj_w + (size_t)half * 256 * 128 + c;
  float acc = 0.f;
  for (int k = 0; k < 256; ++k) acc += nrow[k] * w[(size_t)k * 128];
  (half ? right : left)[(size_t)i * 128 + c] = acc;
}

// ---------------------------------------------------------------------------
// 2. masked pair + pairwise bias:  pair *= m,  pb = (left_i + right_j + b)*m
// ---------------------------------------------------------------------------
__global__ void mask_pb_kernel(const float* __restrict__ pair_in, const float* __restrict__ mask,
                               const float* __restrict__ left, const float* __restrict__ right,
                               const float* __restrict__ proj_b,
                               float* __restrict__ pair_out, float* __restrict__ pb) {
  int t = blockIdx.x * 256 + threadIdx.x;       // LL threads
  if (t >= LL) return;
  int i = t / LSEQ, j = t % LSEQ;
  float m = mask[i] * mask[j];
  const float* pi = pair_in + (size_t)t * CDIM;
  float* po = pair_out + (size_t)t * CDIM;
  float* pp = pb + (size_t)t * CDIM;
  const float* li = left + (size_t)i * CDIM;
  const float* rj = right + (size_t)j * CDIM;
  for (int c = 0; c < CDIM; c += 4) {
    float4 p = *(const float4*)(pi + c);
    float4 l = *(const float4*)(li + c);
    float4 r = *(const float4*)(rj + c);
    float4 b = *(const float4*)(proj_b + c);
    float4 o; o.x = p.x*m; o.y = p.y*m; o.z = p.z*m; o.w = p.w*m;
    float4 q; q.x = (l.x+r.x+b.x)*m; q.y = (l.y+r.y+b.y)*m;
    q.z = (l.z+r.z+b.z)*m; q.w = (l.w+r.w+b.w)*m;
    *(float4*)(po + c) = o;
    *(float4*)(pp + c) = q;
  }
}

// ---------------------------------------------------------------------------
// 3. LayerNorm over 128 channels -> bf16 (wave per row; optional transposed read)
// ---------------------------------------------------------------------------
__global__ void ln_bf16_kernel(const float* __restrict__ src, unsigned short* __restrict__ dst,
                               const float* __restrict__ gain, const float* __restrict__ bias,
                               int transpose) {
  int wid = (blockIdx.x * blockDim.x + threadIdx.x) >> 5;
  int lane = threadIdx.x & 31;
  if (wid >= LL) return;
  int a0 = wid / LSEQ, a1 = wid % LSEQ;
  int srow = transpose ? (a1 * LSEQ + a0) : wid;
  const float* row = src + (size_t)srow * CDIM;
  float4 x = *(const float4*)(row + lane * 4);
  float s = x.x + x.y + x.z + x.w;
  for (int m = 16; m; m >>= 1) s += __shfl_xor(s, m, 32);
  float mean = s * (1.f / 128.f);
  float dx = x.x - mean, dy = x.y - mean, dz = x.z - mean, dw = x.w - mean;
  float vs = dx*dx + dy*dy + dz*dz + dw*dw;
  for (int m = 16; m; m >>= 1) vs += __shfl_xor(vs, m, 32);
  float inv = rsqrtf(vs * (1.f / 128.f) + 1e-5f);
  int c = lane * 4;
  float4 g = *(const float4*)(gain + c);
  float4 b = *(const float4*)(bias + c);
  ushort4 o;
  o.x = f2bf(dx * inv * g.x + b.x);
  o.y = f2bf(dy * inv * g.y + b.y);
  o.z = f2bf(dz * inv * g.z + b.z);
  o.w = f2bf(dw * inv * g.w + b.w);
  *(ushort4*)(dst + (size_t)wid * CDIM + c) = o;
}

// ---------------------------------------------------------------------------
// 4. b2 = bz @ wb  (N=8 per head: too narrow for WMMA; 0.3 GFLOP VALU)
// ---------------------------------------------------------------------------
__global__ void b2_kernel(const unsigned short* __restrict__ bz, const float* __restrict__ wb,
                          float* __restrict__ b2out) {
  size_t t = (size_t)blockIdx.x * 256 + threadIdx.x;   // LL*8
  if (t >= (size_t)LL * NHEAD) return;
  int h = (int)(t & 7);
  size_t r = t >> 3;
  const unsigned short* row = bz + r * CDIM;
  float acc = 0.f;
  for (int c = 0; c < CDIM; ++c) acc += bf2f(row[c]) * wb[(size_t)c * NHEAD + h];
  b2out[t] = acc;
}

// ---------------------------------------------------------------------------
// 5. weight prep: bf16, transposed to [n][k], q-scale / k-norm folded in.
// ---------------------------------------------------------------------------
__global__ void prep_wcat_kernel(const float* __restrict__ wq, const float* __restrict__ wk,
                                 const float* __restrict__ wv, const float* __restrict__ wg,
                                 unsigned short* __restrict__ wcatT) {
  int t = blockIdx.x * 256 + threadIdx.x;      // 512*128
  if (t >= 512 * 128) return;
  int n = t >> 7, k = t & 127;
  int sec = n >> 7, c = n & 127;
  float v;
  if (sec == 0)      v = wq[(size_t)k * 128 + c] * SCALE_Q;
  else if (sec == 1) v = wk[(size_t)k * 128 + c] * INV_L;
  else if (sec == 2) v = wv[(size_t)k * 128 + c];
  else               v = wg[(size_t)k * 128 + c];
  wcatT[(size_t)n * 128 + k] = f2bf(v);
}

__global__ void prep_w128_kernel(const float* __restrict__ w, unsigned short* __restrict__ wT) {
  int t = blockIdx.x * 256 + threadIdx.x;      // 128*128
  if (t >= 128 * 128) return;
  int n = t >> 7, k = t & 127;
  wT[(size_t)n * 128 + k] = f2bf(w[(size_t)k * 128 + n]);
}

// ---------------------------------------------------------------------------
// 6. fused q/k/v/g projection: (LL x 128) @ (128 x 512) via WMMA.
//    Async LDS fill; epilogue: q,k bf16; v -> [h][n][d][j]; g = sigmoid.
// ---------------------------------------------------------------------------
#define LDSA_STRIDE 136   // 128 + 8 pad halves (272B; conflict-free)
__global__ __launch_bounds__(256) void proj_qkvg_kernel(
    const unsigned short* __restrict__ xbf, const unsigned short* __restrict__ wcatT,
    const float* __restrict__ bg,
    unsigned short* __restrict__ qbf, unsigned short* __restrict__ kbf,
    unsigned short* __restrict__ vT,  unsigned short* __restrict__ gbf) {
  __shared__ __align__(16) unsigned short lA[64 * LDSA_STRIDE];
  __shared__ __align__(16) unsigned short lB[128 * LDSA_STRIDE];
  const int m0 = blockIdx.x * 64;
  const int ntile = blockIdx.y;                 // 0..3
  const int tid = threadIdx.x;

  for (int f = tid; f < 1024; f += 256)
    async_b128(lds_off32(&lA[(f >> 4) * LDSA_STRIDE + (f & 15) * 8]),
               xbf + (size_t)m0 * CDIM + (size_t)f * 8);
  for (int f = tid; f < 2048; f += 256)
    async_b128(lds_off32(&lB[(f >> 4) * LDSA_STRIDE + (f & 15) * 8]),
               wcatT + (size_t)ntile * 128 * 128 + (size_t)f * 8);
  asm volatile("s_wait_asynccnt 0x0" ::: "memory");
  __syncthreads();

  const int wv_ = tid >> 5, lane = tid & 31;
  const int msub = wv_ & 3, ngrp = wv_ >> 2;
  const int lan15 = lane & 15;
  v8f zero = {0.f,0.f,0.f,0.f,0.f,0.f,0.f,0.f};
  v8f acc[4] = {zero, zero, zero, zero};
  const unsigned short* arow = lA + (msub * 16 + lan15) * LDSA_STRIDE;
  for (int ks = 0; ks < 4; ++ks) {
    v16bf a = load_fragA(arow + ks * 32, lane);
    #pragma unroll
    for (int ns = 0; ns < 4; ++ns) {
      const unsigned short* brow = lB + ((ngrp * 4 + ns) * 16 + lan15) * LDSA_STRIDE;
      acc[ns] = wmma_bf16(a, load_fragB(brow + ks * 32, lane), acc[ns]);
    }
  }
  const int mhi = (lane & 16) ? 8 : 0;
  for (int ns = 0; ns < 4; ++ns) {
    int nc = ntile * 128 + (ngrp * 4 + ns) * 16 + lan15;
    int sec = nc >> 7, c = nc & 127;
    AccU u; u.v = acc[ns];
    #pragma unroll
    for (int e = 0; e < 8; ++e) {
      int r = m0 + msub * 16 + e + mhi;
      float val = u.f[e];
      if (sec == 0) {
        qbf[(size_t)r * CDIM + c] = f2bf(val);
      } else if (sec == 1) {
        kbf[(size_t)r * CDIM + c] = f2bf(val);
      } else if (sec == 2) {
        int n = r / LSEQ, j = r % LSEQ, h = c >> 4, d = c & 15;
        vT[((size_t)(h * LSEQ + n) * DHID + d) * LSEQ + j] = f2bf(val);
      } else {
        float gv = 1.f / (1.f + __expf(-(val + bg[c])));
        gbf[(size_t)r * CDIM + c] = f2bf(gv);
      }
    }
  }
}

// ---------------------------------------------------------------------------
// 7. tied-attention logits: attn[h,i,j] = sum_{n,d} q[n,i,h,d]*k[n,j,h,d]
//    + b2[i,j,h], masked.  64x64 tile; K loop of 32 (two n-slices of 16 d),
//    double-buffered LDS fed by async loads, 2 tiles in flight.
// ---------------------------------------------------------------------------
#define LG_STRIDE 40      // 32 + 8 pad halves (80B)
__global__ __launch_bounds__(256) void logits_kernel(
    const unsigned short* __restrict__ qbf, const unsigned short* __restrict__ kbf,
    const float* __restrict__ b2, const float* __restrict__ mask,
    float* __restrict__ attnL) {
  __shared__ __align__(16) unsigned short lA[2][64 * LG_STRIDE];
  __shared__ __align__(16) unsigned short lB[2][64 * LG_STRIDE];
  const int i0 = blockIdx.x * 64, j0 = blockIdx.y * 64, h = blockIdx.z;
  const int tid = threadIdx.x, wv_ = tid >> 5, lane = tid & 31;
  const int isub = wv_ & 3, jgrp = wv_ >> 2;
  const int lan15 = lane & 15;
  v8f zero = {0.f,0.f,0.f,0.f,0.f,0.f,0.f,0.f};
  v8f acc[2] = {zero, zero};

  const int i_ld = tid >> 2, seg = tid & 3;     // 256 x 16B per tile half
  const int nn = seg >> 1, d0 = (seg & 1) * 8;
  unsigned ldsA[2], ldsB[2];
  ldsA[0] = lds_off32(&lA[0][i_ld * LG_STRIDE + seg * 8]);
  ldsA[1] = lds_off32(&lA[1][i_ld * LG_STRIDE + seg * 8]);
  ldsB[0] = lds_off32(&lB[0][i_ld * LG_STRIDE + seg * 8]);
  ldsB[1] = lds_off32(&lB[1][i_ld * LG_STRIDE + seg * 8]);

  auto issue = [&](int t) {
    const unsigned short* ga =
        qbf + ((size_t)((2 * t + nn) * LSEQ + i0 + i_ld)) * CDIM + h * DHID + d0;
    const unsigned short* gb =
        kbf + ((size_t)((2 * t + nn) * LSEQ + j0 + i_ld)) * CDIM + h * DHID + d0;
    async_b128(ldsA[t & 1], ga);    // 1 async instr per wave
    async_b128(ldsB[t & 1], gb);    // 1 async instr per wave
  };

  const int NT = LSEQ / 2;          // 192 K-tiles
  issue(0);
  issue(1);
  for (int t = 0; t < NT; ++t) {
    // retire tile t: 2 async instrs outstanding iff tile t+1 already issued
    if (t + 1 < NT) asm volatile("s_wait_asynccnt 0x2" ::: "memory");
    else            asm volatile("s_wait_asynccnt 0x0" ::: "memory");
    __syncthreads();                // everyone's tile-t data visible
    const int buf = t & 1;
    v16bf a = load_fragA(&lA[buf][(isub * 16 + lan15) * LG_STRIDE], lane);
    #pragma unroll
    for (int t2 = 0; t2 < 2; ++t2) {
      const unsigned short* brow = &lB[buf][((jgrp * 2 + t2) * 16 + lan15) * LG_STRIDE];
      acc[t2] = wmma_bf16(a, load_fragB(brow, lane), acc[t2]);
    }
    __syncthreads();                // all reads of buf done before overwrite
    if (t + 2 < NT) issue(t + 2);
  }
  const int mhi = (lane & 16) ? 8 : 0;
  for (int t2 = 0; t2 < 2; ++t2) {
    int j = j0 + (jgrp * 2 + t2) * 16 + lan15;
    float mj = mask[j];
    AccU u; u.v = acc[t2];
    #pragma unroll
    for (int e = 0; e < 8; ++e) {
      int i = i0 + isub * 16 + e + mhi;
      float v = u.f[e] + b2[((size_t)i * LSEQ + j) * NHEAD + h];
      if (!(mj > 0.f)) v = -1e9f;
      attnL[(size_t)h * LL + (size_t)i * LSEQ + j] = v;
    }
  }
}

// ---------------------------------------------------------------------------
// 8. softmax over j (one wave per (h,i) row of 384), output bf16
// ---------------------------------------------------------------------------
__global__ void softmax_kernel(const float* __restrict__ attnL, unsigned short* __restrict__ attnbf) {
  int row = (blockIdx.x * blockDim.x + threadIdx.x) >> 5;    // h*384+i
  int lane = threadIdx.x & 31;
  if (row >= NHEAD * LSEQ) return;
  const float* src = attnL + (size_t)row * LSEQ;
  float v[12]; float mx = -1e30f;
  #pragma unroll
  for (int k = 0; k < 12; ++k) { v[k] = src[lane + 32 * k]; mx = fmaxf(mx, v[k]); }
  for (int m = 16; m; m >>= 1) mx = fmaxf(mx, __shfl_xor(mx, m, 32));
  float s = 0.f;
  #pragma unroll
  for (int k = 0; k < 12; ++k) { v[k] = __expf(v[k] - mx); s += v[k]; }
  for (int m = 16; m; m >>= 1) s += __shfl_xor(s, m, 32);
  float inv = 1.f / s;
  unsigned short* dst = attnbf + (size_t)row * LSEQ;
  #pragma unroll
  for (int k = 0; k < 12; ++k) dst[lane + 32 * k] = f2bf(v[k] * inv);
}

// ---------------------------------------------------------------------------
// 9. AV + gating: out[n,i,h,d] = g * sum_j attn[h,i,j] v[n,j,h,d]
//    attn tile async-staged to LDS and cached in register fragments;
//    v fragments loaded contiguously from the [h][n][d][j] transpose.
// ---------------------------------------------------------------------------
#define AV_STRIDE 392     // 384 + 8 pad halves (784B)
__global__ __launch_bounds__(256) void av_kernel(
    const unsigned short* __restrict__ attnbf, const unsigned short* __restrict__ vT,
    const unsigned short* __restrict__ gbf, unsigned short* __restrict__ gav) {
  __shared__ __align__(16) unsigned short lA[64 * AV_STRIDE];
  const int nchunk = blockIdx.x, i0 = blockIdx.y * 64, h = blockIdx.z;
  const int tid = threadIdx.x;
  for (int f = tid; f < 3072; f += 256) {
    int i = f / 48, seg = f % 48;
    async_b128(lds_off32(&lA[i * AV_STRIDE + seg * 8]),
               attnbf + ((size_t)(h * LSEQ + i0 + i)) * LSEQ + seg * 8);
  }
  asm volatile("s_wait_asynccnt 0x0" ::: "memory");
  __syncthreads();
  const int wv_ = tid >> 5, lane = tid & 31;
  const int isub = wv_ & 3, nhalf = wv_ >> 2;
  const int lan15 = lane & 15;
  const unsigned short* arow = lA + (isub * 16 + lan15) * AV_STRIDE;
  v16bf afr[12];
  #pragma unroll
  for (int ks = 0; ks < 12; ++ks) afr[ks] = load_fragA(arow + ks * 32, lane);
  const int d = lan15;
  const int kb = (lane & 16) ? 16 : 0;
  const int mhi = (lane & 16) ? 8 : 0;
  v8f zero = {0.f,0.f,0.f,0.f,0.f,0.f,0.f,0.f};
  for (int nn = 0; nn < 24; ++nn) {
    int n = nchunk * 48 + nhalf * 24 + nn;
    const unsigned short* vrow = vT + ((size_t)(h * LSEQ + n) * DHID + d) * LSEQ;
    v8f acc = zero;
    #pragma unroll
    for (int ks = 0; ks < 12; ++ks) {
      FragBF fb;
      fb.q[0] = *(const uint4*)(vrow + ks * 32 + kb);
      fb.q[1] = *(const uint4*)(vrow + ks * 32 + kb + 8);
      acc = wmma_bf16(afr[ks], fb.v, acc);
    }
    AccU u; u.v = acc;
    #pragma unroll
    for (int e = 0; e < 8; ++e) {
      int i = i0 + isub * 16 + e + mhi;
      size_t idx = ((size_t)(n * LSEQ + i)) * CDIM + h * DHID + d;
      gav[idx] = f2bf(bf2f(gbf[idx]) * u.f[e]);
    }
  }
}

// ---------------------------------------------------------------------------
// 10. 128x128 WMMA GEMM (async LDS fill), epilogue modes:
//     mode 0: y = acc + bias;  pair[maybe-transposed row] += y  (out proj)
//     mode 1: y = relu(acc + bias); store bf16                  (FF layer 1)
//     mode 2: y = acc + bias;  pair[row] += y                   (FF layer 2)
// ---------------------------------------------------------------------------
__global__ __launch_bounds__(256) void gemm128_kernel(
    const unsigned short* __restrict__ Abf, const unsigned short* __restrict__ BT,
    const float* __restrict__ biasv, float* __restrict__ pairAcc,
    unsigned short* __restrict__ dstbf, int mode, int transpose) {
  __shared__ __align__(16) unsigned short lA[64 * LDSA_STRIDE];
  __shared__ __align__(16) unsigned short lB[128 * LDSA_STRIDE];
  const int m0 = blockIdx.x * 64;
  const int tid = threadIdx.x;
  for (int f = tid; f < 1024; f += 256)
    async_b128(lds_off32(&lA[(f >> 4) * LDSA_STRIDE + (f & 15) * 8]),
               Abf + (size_t)m0 * CDIM + (size_t)f * 8);
  for (int f = tid; f < 2048; f += 256)
    async_b128(lds_off32(&lB[(f >> 4) * LDSA_STRIDE + (f & 15) * 8]),
               BT + (size_t)f * 8);
  asm volatile("s_wait_asynccnt 0x0" ::: "memory");
  __syncthreads();
  const int wv_ = tid >> 5, lane = tid & 31;
  const int msub = wv_ & 3, ngrp = wv_ >> 2;
  const int lan15 = lane & 15;
  v8f zero = {0.f,0.f,0.f,0.f,0.f,0.f,0.f,0.f};
  v8f acc[4] = {zero, zero, zero, zero};
  const unsigned short* arow = lA + (msub * 16 + lan15) * LDSA_STRIDE;
  for (int ks = 0; ks < 4; ++ks) {
    v16bf a = load_fragA(arow + ks * 32, lane);
    #pragma unroll
    for (int ns = 0; ns < 4; ++ns) {
      const unsigned short* brow = lB + ((ngrp * 4 + ns) * 16 + lan15) * LDSA_STRIDE;
      acc[ns] = wmma_bf16(a, load_fragB(brow + ks * 32, lane), acc[ns]);
    }
  }
  const int mhi = (lane & 16) ? 8 : 0;
  for (int ns = 0; ns < 4; ++ns) {
    int c = (ngrp * 4 + ns) * 16 + lan15;
    float bc = biasv[c];
    AccU u; u.v = acc[ns];
    #pragma unroll
    for (int e = 0; e < 8; ++e) {
      int r = m0 + msub * 16 + e + mhi;
      float y = u.f[e] + bc;
      if (mode == 1) {
        dstbf[(size_t)r * CDIM + c] = f2bf(fmaxf(y, 0.f));
      } else {
        size_t pr = (size_t)r;
        if (transpose) { int a0 = r / LSEQ, a1 = r % LSEQ; pr = (size_t)a1 * LSEQ + a0; }
        pairAcc[pr * CDIM + c] += y;
      }
    }
  }
}

// ---------------------------------------------------------------------------
// host orchestration
// ---------------------------------------------------------------------------
extern "C" void kernel_launch(void* const* d_in, const int* in_sizes, int n_in,
                              void* d_out, int out_size, void* d_ws, size_t ws_size,
                              hipStream_t stream) {
  (void)in_sizes; (void)n_in; (void)out_size; (void)ws_size;
  const float* node   = (const float*)d_in[0];
  const float* pair   = (const float*)d_in[1];
  const float* mask   = (const float*)d_in[2];
  const float* proj_w = (const float*)d_in[3];
  const float* proj_b = (const float*)d_in[4];
  const float* rowp[12]; for (int i = 0; i < 12; ++i) rowp[i] = (const float*)d_in[5 + i];
  const float* colp[12]; for (int i = 0; i < 12; ++i) colp[i] = (const float*)d_in[17 + i];
  const float* ff_ln_g = (const float*)d_in[29];
  const float* ff_ln_b = (const float*)d_in[30];
  const float* ff_w1   = (const float*)d_in[31];
  const float* ff_b1   = (const float*)d_in[32];
  const float* ff_w2   = (const float*)d_in[33];
  const float* ff_b2   = (const float*)d_in[34];
  float* outp = (float*)d_out;

  // workspace carve-out (~310 MB)
  char* ws = (char*)d_ws; size_t off = 0;
  auto carve = [&](size_t bytes) -> void* {
    void* p = ws + off; off = (off + bytes + 255) & ~(size_t)255; return p;
  };
  float* pb            = (float*)carve((size_t)LL * CDIM * 4);
  unsigned short* xbf  = (unsigned short*)carve((size_t)LL * CDIM * 2);
  unsigned short* bzbf = (unsigned short*)carve((size_t)LL * CDIM * 2); // reused as gated-AV buf
  unsigned short* qbf  = (unsigned short*)carve((size_t)LL * CDIM * 2); // reused as FF hidden buf
  unsigned short* kbf  = (unsigned short*)carve((size_t)LL * CDIM * 2);
  unsigned short* vTbf = (unsigned short*)carve((size_t)LL * CDIM * 2);
  unsigned short* gbf  = (unsigned short*)carve((size_t)LL * CDIM * 2);
  float* b2buf         = (float*)carve((size_t)LL * NHEAD * 4);
  float* attnL         = (float*)carve((size_t)NHEAD * LL * 4);
  unsigned short* atbf = (unsigned short*)carve((size_t)NHEAD * LL * 2);
  float* left          = (float*)carve((size_t)LSEQ * CDIM * 4);
  float* right         = (float*)carve((size_t)LSEQ * CDIM * 4);
  unsigned short* wcatT = (unsigned short*)carve((size_t)512 * 128 * 2);
  unsigned short* wT    = (unsigned short*)carve((size_t)128 * 128 * 2);

  // node bias projection + masked pair / pb
  node_proj_kernel<<<LSEQ * 256 / 256, 256, 0, stream>>>(node, proj_w, left, right);
  mask_pb_kernel<<<LL / 256, 256, 0, stream>>>(pair, mask, left, right, proj_b, outp, pb);

  const int lnBlocks = LL / 8;   // 8 waves per 256-thread block

  auto run_attn = [&](const float* const* P, int transpose) {
    const float *ln_pg = P[0], *ln_pb_ = P[1], *ln_bg = P[2], *ln_bb = P[3];
    const float *wq = P[4], *wk = P[5], *wvv = P[6], *wb = P[7], *wg = P[8];
    const float *bg = P[9], *wo = P[10], *bo = P[11];
    prep_wcat_kernel<<<512 * 128 / 256, 256, 0, stream>>>(wq, wk, wvv, wg, wcatT);
    prep_w128_kernel<<<128 * 128 / 256, 256, 0, stream>>>(wo, wT);
    ln_bf16_kernel<<<lnBlocks, 256, 0, stream>>>(outp, xbf, ln_pg, ln_pb_, transpose);
    ln_bf16_kernel<<<lnBlocks, 256, 0, stream>>>(pb, bzbf, ln_bg, ln_bb, transpose);
    b2_kernel<<<(LL * NHEAD) / 256, 256, 0, stream>>>(bzbf, wb, b2buf);
    proj_qkvg_kernel<<<dim3(LL / 64, 4), 256, 0, stream>>>(xbf, wcatT, bg, qbf, kbf, vTbf, gbf);
    logits_kernel<<<dim3(LSEQ / 64, LSEQ / 64, NHEAD), 256, 0, stream>>>(qbf, kbf, b2buf, mask, attnL);
    softmax_kernel<<<(NHEAD * LSEQ) / 8, 256, 0, stream>>>(attnL, atbf);
    av_kernel<<<dim3(8, LSEQ / 64, NHEAD), 256, 0, stream>>>(atbf, vTbf, gbf, bzbf /*gated AV*/);
    gemm128_kernel<<<LL / 64, 256, 0, stream>>>(bzbf, wT, bo, outp, nullptr, 0, transpose);
  };

  run_attn(rowp, 1);   // row attention (i<->j transposed view)
  run_attn(colp, 0);   // col attention

  // feed-forward
  ln_bf16_kernel<<<lnBlocks, 256, 0, stream>>>(outp, xbf, ff_ln_g, ff_ln_b, 0);
  prep_w128_kernel<<<128 * 128 / 256, 256, 0, stream>>>(ff_w1, wT);
  gemm128_kernel<<<LL / 64, 256, 0, stream>>>(xbf, wT, ff_b1, outp, qbf, 1, 0);   // relu -> bf16
  prep_w128_kernel<<<128 * 128 / 256, 256, 0, stream>>>(ff_w2, wT);
  gemm128_kernel<<<LL / 64, 256, 0, stream>>>(qbf, wT, ff_b2, outp, nullptr, 2, 0); // += residual
}
